// Multi_Head_Attention_64553358458924
// MI455X (gfx1250) — compile-verified, matrix operand order checked by reading, not run on hardware
//
#include <hip/hip_runtime.h>
#include <hip/hip_bf16.h>

// ---------------------------------------------------------------------------
// MHA forward, B=2, S=4096, D=512, H=8, Dh=64 on gfx1250 (CDNA5).
// bf16 WMMA everywhere; async global->LDS staging with double buffering;
// all transposes hoisted out of hot loops (pre-transposed weights / V).
// ---------------------------------------------------------------------------

typedef __attribute__((ext_vector_type(16))) __bf16 v16bf;
typedef __attribute__((ext_vector_type(8)))  float  v8f;

union FragAB { v16bf v; uint4 q[2]; };
union FragC  { v8f   v; float f[8]; };

#define BATCH   2
#define SLEN    4096
#define DMODEL  512
#define NHEAD   8
#define HD      64
#define QKV_LD  1536   // Q|K|V concatenated columns

// LDS pitches (elements); rows stay 16B aligned
#define AP 40   // 32-wide K-slab tiles + 8 pad (80 B rows)
#define KP 72   // 64-wide tiles + 8 pad (144 B rows)
#define PP 72

__device__ __forceinline__ unsigned short f2bf(float f) {
    unsigned u = __builtin_bit_cast(unsigned, f);
    unsigned r = (u + 0x7FFFu + ((u >> 16) & 1u)) >> 16;   // RNE
    return (unsigned short)r;
}

// CDNA5 16-bit WMMA A/B operand: lane (i, half) holds K-runs
//   elements 0..7  <- K = 8*half .. 8*half+7
//   elements 8..15 <- K = 16+8*half .. 23+8*half
__device__ __forceinline__ FragAB ldfrag(const unsigned short* base, int half) {
    FragAB f;
    f.q[0] = *(const uint4*)(base + 8 * half);
    f.q[1] = *(const uint4*)(base + 16 + 8 * half);
    return f;
}

__device__ __forceinline__ v8f wmma_bf16(const FragAB& a, const FragAB& b, v8f c) {
    return __builtin_amdgcn_wmma_f32_16x16x32_bf16(
        false, a.v, false, b.v, (short)0, c, false, false);
}

// async DMA: 16 B per lane, global -> LDS, tracked by ASYNCcnt
__device__ __forceinline__ void async_b128(unsigned lds_off, const void* gaddr) {
    asm volatile("global_load_async_to_lds_b128 %0, %1, off"
                 :: "v"(lds_off), "v"(gaddr) : "memory");
}
__device__ __forceinline__ void wait_async0() {
    asm volatile("s_wait_asynccnt 0x0" ::: "memory");
}
// low 32 bits of the shared-aperture generic address == LDS byte offset
__device__ __forceinline__ unsigned lds_off(const void* p) {
    return (unsigned)(size_t)p;
}

// ---------------------------------------------------------------------------
// conversion / packing kernels (one-time, tiny)
// ---------------------------------------------------------------------------
__global__ __launch_bounds__(256) void cvt_f32_bf16(const float* __restrict__ src,
                                                    unsigned short* __restrict__ dst,
                                                    int n) {
    int i = (blockIdx.x * blockDim.x + threadIdx.x) * 4;
    if (i >= n) return;
    float4 f = *(const float4*)(src + i);
    unsigned a = f2bf(f.x), b = f2bf(f.y), c = f2bf(f.z), d = f2bf(f.w);
    uint2 o; o.x = a | (b << 16); o.y = c | (d << 16);
    *(uint2*)(dst + i) = o;
}

// pack [Wq|Wk|Wv] transposed: dst[n][k] (n in 0..1535) = W(sec)[k][n%512], bf16
__global__ __launch_bounds__(256) void pack_wqkvT(const float* __restrict__ Wq,
                                                  const float* __restrict__ Wk,
                                                  const float* __restrict__ Wv,
                                                  unsigned short* __restrict__ dst) {
    int i = blockIdx.x * blockDim.x + threadIdx.x;   // over 512*512
    int k = i >> 9, n = i & 511;                     // coalesced reads along n
    dst[(size_t)n * DMODEL + k]          = f2bf(Wq[i]);
    dst[(size_t)(512 + n) * DMODEL + k]  = f2bf(Wk[i]);
    dst[(size_t)(1024 + n) * DMODEL + k] = f2bf(Wv[i]);
}

__global__ __launch_bounds__(256) void pack_woT(const float* __restrict__ Wo,
                                                unsigned short* __restrict__ dst) {
    int i = blockIdx.x * blockDim.x + threadIdx.x;
    int k = i >> 9, n = i & 511;
    dst[(size_t)n * DMODEL + k] = f2bf(Wo[i]);
}

// transpose the V section of qkv into vt[B*H*Dh][S] (bf16) via LDS tiles
__global__ __launch_bounds__(256) void transpose_v(const unsigned short* __restrict__ qkv,
                                                   unsigned short* __restrict__ vt) {
    __shared__ __align__(16) unsigned short tile[64 * KP];   // [s][d]
    const int t  = threadIdx.x;
    const int bh = blockIdx.y;
    const int b  = bh >> 3, h = bh & 7;
    const int s0 = blockIdx.x * 64;
    const int r  = t >> 3;          // 0..31
    const int c  = (t & 7) * 8;     // 0..56

    const unsigned short* src = qkv + ((size_t)(b * SLEN + s0)) * QKV_LD + 1024 + h * HD;
    *(uint4*)&tile[r * KP + c]        = *(const uint4*)(src + (size_t)r * QKV_LD + c);
    *(uint4*)&tile[(r + 32) * KP + c] = *(const uint4*)(src + (size_t)(r + 32) * QKV_LD + c);
    __syncthreads();

    union { uint4 q; unsigned short s16[8]; } o0, o1;
#pragma unroll
    for (int i = 0; i < 8; ++i) {
        o0.s16[i] = tile[(c + i) * KP + r];        // d = r
        o1.s16[i] = tile[(c + i) * KP + r + 32];   // d = r+32
    }
    unsigned short* drow0 = vt + ((size_t)bh * HD + r) * SLEN + s0 + c;
    unsigned short* drow1 = vt + ((size_t)bh * HD + r + 32) * SLEN + s0 + c;
    *(uint4*)drow0 = o0.q;
    *(uint4*)drow1 = o1.q;
}

// ---------------------------------------------------------------------------
// bf16 GEMM (TN): C[M,N] = A[M,K] * Bt[N,K]^T.  Block tile 128x128, 8 waves,
// wave tile 32x64 (8 WMMA / K-step, B-fragment reuse). Async double-buffered.
// ---------------------------------------------------------------------------
__global__ __launch_bounds__(256) void gemm_bf16_tn(const unsigned short* __restrict__ A,
                                                    const unsigned short* __restrict__ Bt,
                                                    void* __restrict__ C,
                                                    int M, int N, int K, int outBf16) {
    __shared__ __align__(16) unsigned short Al[2][128 * AP];
    __shared__ __align__(16) unsigned short Bl[2][128 * AP];

    const int t    = threadIdx.x;
    const int w    = t >> 5;
    const int L    = t & 31;
    const int lm   = L & 15;
    const int half = L >> 4;
    const int wm0  = (w & 3) * 32;
    const int wn0  = (w >> 2) * 64;
    const int m0   = blockIdx.x * 128;
    const int n0   = blockIdx.y * 128;

    FragC acc[2][4];
#pragma unroll
    for (int i = 0; i < 2; ++i)
#pragma unroll
        for (int tn = 0; tn < 4; ++tn) acc[i][tn].v = v8f{};

    // staging: both tiles are 128 rows x 32 cols; 2 b128 copies per tile/thread
    const int srow = t >> 2;          // 0..63 (and +64)
    const int scol = (t & 3) * 8;     // 0,8,16,24
    const unsigned short* Ag0 = A  + (size_t)(m0 + srow) * K + scol;
    const unsigned short* Ag1 = A  + (size_t)(m0 + srow + 64) * K + scol;
    const unsigned short* Bg0 = Bt + (size_t)(n0 + srow) * K + scol;
    const unsigned short* Bg1 = Bt + (size_t)(n0 + srow + 64) * K + scol;

    auto stage = [&](int buf, int kk) {
        async_b128(lds_off(&Al[buf][srow * AP + scol]),        Ag0 + kk);
        async_b128(lds_off(&Al[buf][(srow + 64) * AP + scol]), Ag1 + kk);
        async_b128(lds_off(&Bl[buf][srow * AP + scol]),        Bg0 + kk);
        async_b128(lds_off(&Bl[buf][(srow + 64) * AP + scol]), Bg1 + kk);
    };

    stage(0, 0);
    wait_async0();
    __syncthreads();

    int buf = 0;
    for (int kk = 0; kk < K; kk += 32) {
        if (kk + 32 < K) stage(buf ^ 1, kk + 32);

        FragAB a0 = ldfrag(&Al[buf][(wm0 + lm) * AP], half);
        FragAB a1 = ldfrag(&Al[buf][(wm0 + 16 + lm) * AP], half);
#pragma unroll
        for (int tn = 0; tn < 4; ++tn) {
            FragAB bf = ldfrag(&Bl[buf][(wn0 + tn * 16 + lm) * AP], half);
            acc[0][tn].v = wmma_bf16(a0, bf, acc[0][tn].v);
            acc[1][tn].v = wmma_bf16(a1, bf, acc[1][tn].v);
        }
        wait_async0();
        __syncthreads();
        buf ^= 1;
    }

#pragma unroll
    for (int i = 0; i < 2; ++i)
#pragma unroll
        for (int tn = 0; tn < 4; ++tn)
#pragma unroll
            for (int v = 0; v < 8; ++v) {
                size_t row = (size_t)m0 + wm0 + i * 16 + v + 8 * half;
                size_t col = (size_t)n0 + wn0 + tn * 16 + lm;
                if (outBf16) ((unsigned short*)C)[row * N + col] = f2bf(acc[i][tn].f[v]);
                else         ((float*)C)[row * N + col]          = acc[i][tn].f[v];
            }
}

// ---------------------------------------------------------------------------
// Flash attention: block = (b,h) x 128 q-rows; wave owns 16 q-rows.
// 64-key blocks, async double-buffered K / V^T staging, 16 WMMA per block.
// ---------------------------------------------------------------------------
__global__ __launch_bounds__(256) void attn_fwd(const unsigned short* __restrict__ qkv,
                                                const unsigned short* __restrict__ vt,
                                                unsigned short* __restrict__ ctx) {
    __shared__ __align__(16) unsigned short Kl[2][64 * KP];   // [key][dh]
    __shared__ __align__(16) unsigned short Vl[2][64 * KP];   // [dh][key] (V^T)
    __shared__ __align__(16) unsigned short Pl[8 * 16 * PP];  // per-wave P scratch

    const int t    = threadIdx.x;
    const int w    = t >> 5;
    const int L    = t & 31;
    const int lm   = L & 15;
    const int half = L >> 4;
    const int bh   = blockIdx.y;
    const int b    = bh >> 3;
    const int h    = bh & 7;
    const int q0   = blockIdx.x * 128 + w * 16;

    // Q fragments pinned in registers (dh 0..31 / 32..63)
    const unsigned short* qbase =
        qkv + ((size_t)b * SLEN + q0 + lm) * QKV_LD + h * HD;
    FragAB qa0 = ldfrag(qbase, half);
    FragAB qa1 = ldfrag(qbase + 32, half);

    FragC o[4];
#pragma unroll
    for (int tn = 0; tn < 4; ++tn) o[tn].v = v8f{};
    float mrun[8], lrun[8];
#pragma unroll
    for (int v = 0; v < 8; ++v) { mrun[v] = -3.0e38f; lrun[v] = 0.0f; }

    // staging bases: 64x64 tiles, 2 b128 copies per tile per thread
    const int srow = t >> 2;          // 0..63  (key for K, dh for V^T)
    const int scol = (t & 3) * 8;     // 0,8,16,24
    const unsigned short* kg0 = qkv + ((size_t)(b * SLEN) + srow) * QKV_LD + 512 + h * HD + scol;
    const unsigned short* vg0 = vt + ((size_t)bh * HD + srow) * SLEN + scol;

    auto stage = [&](int buf, int j) {
        const unsigned short* kg = kg0 + (size_t)j * QKV_LD;
        const unsigned short* vg = vg0 + j;
        async_b128(lds_off(&Kl[buf][srow * KP + scol]),      kg);
        async_b128(lds_off(&Kl[buf][srow * KP + scol + 32]), kg + 32);
        async_b128(lds_off(&Vl[buf][srow * KP + scol]),      vg);
        async_b128(lds_off(&Vl[buf][srow * KP + scol + 32]), vg + 32);
    };

    stage(0, 0);
    wait_async0();
    __syncthreads();

    unsigned short* pw = &Pl[w * 16 * PP];
    int buf = 0;
    for (int j = 0; j < SLEN; j += 64) {
        if (j + 64 < SLEN) stage(buf ^ 1, j + 64);

        // scores: 4 key tiles x (Dh=64 -> 2 K-steps)
        FragC sc[4];
#pragma unroll
        for (int kt = 0; kt < 4; ++kt) {
            sc[kt].v = v8f{};
            const unsigned short* kb = &Kl[buf][(kt * 16 + lm) * KP];
            FragAB b0 = ldfrag(kb, half);
            FragAB b1 = ldfrag(kb + 32, half);
            sc[kt].v = wmma_bf16(qa0, b0, sc[kt].v);
            sc[kt].v = wmma_bf16(qa1, b1, sc[kt].v);
        }

        // online softmax (row = v + 8*half; cols across the 16-lane group)
        float p[4][8], alpha[8];
#pragma unroll
        for (int v = 0; v < 8; ++v) {
            float x0 = sc[0].f[v] * 0.125f, x1 = sc[1].f[v] * 0.125f;
            float x2 = sc[2].f[v] * 0.125f, x3 = sc[3].f[v] * 0.125f;
            float mx = fmaxf(fmaxf(x0, x1), fmaxf(x2, x3));
#pragma unroll
            for (int off = 1; off < 16; off <<= 1)
                mx = fmaxf(mx, __shfl_xor(mx, off, 32));
            float mnew = fmaxf(mrun[v], mx);
            alpha[v] = __expf(mrun[v] - mnew);
            mrun[v]  = mnew;
            p[0][v] = __expf(x0 - mnew);
            p[1][v] = __expf(x1 - mnew);
            p[2][v] = __expf(x2 - mnew);
            p[3][v] = __expf(x3 - mnew);
            float rs = (p[0][v] + p[1][v]) + (p[2][v] + p[3][v]);
#pragma unroll
            for (int off = 1; off < 16; off <<= 1)
                rs += __shfl_xor(rs, off, 32);
            lrun[v] = lrun[v] * alpha[v] + rs;
        }

        // rescale O; spill P (C-layout) -> per-wave LDS for A-layout reload
#pragma unroll
        for (int v = 0; v < 8; ++v) {
#pragma unroll
            for (int tn = 0; tn < 4; ++tn) o[tn].f[v] *= alpha[v];
            unsigned short* pr = &pw[(v + 8 * half) * PP + lm];
#pragma unroll
            for (int kt = 0; kt < 4; ++kt) pr[kt * 16] = f2bf(p[kt][v]);
        }
        asm volatile("s_wait_dscnt 0x0" ::: "memory");   // same-wave LDS RAW fence

        // ctx += P(16x64) * V(64x64): 2 K-steps x 4 dh tiles
        FragAB pa0 = ldfrag(&pw[lm * PP], half);
        FragAB pa1 = ldfrag(&pw[lm * PP] + 32, half);
#pragma unroll
        for (int tn = 0; tn < 4; ++tn) {
            const unsigned short* vb = &Vl[buf][(tn * 16 + lm) * KP];
            FragAB v0 = ldfrag(vb, half);
            FragAB v1 = ldfrag(vb + 32, half);
            o[tn].v = wmma_bf16(pa0, v0, o[tn].v);
            o[tn].v = wmma_bf16(pa1, v1, o[tn].v);
        }

        wait_async0();
        __syncthreads();
        buf ^= 1;
    }

    // normalize and emit ctx (bf16) at [b, q, h*64 + dh]
#pragma unroll
    for (int v = 0; v < 8; ++v) {
        float inv = 1.0f / lrun[v];
        size_t row = (size_t)b * SLEN + q0 + v + 8 * half;
#pragma unroll
        for (int tn = 0; tn < 4; ++tn)
            ctx[row * DMODEL + h * HD + tn * 16 + lm] = f2bf(o[tn].f[v] * inv);
    }
}

// ---------------------------------------------------------------------------
// launch
// ---------------------------------------------------------------------------
extern "C" void kernel_launch(void* const* d_in, const int* in_sizes, int n_in,
                              void* d_out, int out_size, void* d_ws, size_t ws_size,
                              hipStream_t stream) {
    const float* x  = (const float*)d_in[0];
    const float* Wq = (const float*)d_in[1];
    const float* Wk = (const float*)d_in[2];
    const float* Wv = (const float*)d_in[3];
    const float* Wo = (const float*)d_in[4];
    float* out = (float*)d_out;

    char* ws = (char*)d_ws;
    const size_t M = (size_t)BATCH * SLEN;                               // 8192
    unsigned short* x_bf   = (unsigned short*)(ws);                      // 8 MB
    unsigned short* wqkvT  = (unsigned short*)(ws + 8388608);            // 1.5 MB
    unsigned short* woT    = (unsigned short*)(ws + 9961472);            // 0.5 MB
    unsigned short* qkv    = (unsigned short*)(ws + 10485760);           // 24 MB
    unsigned short* vtb    = (unsigned short*)(ws + 35651584);           // 8 MB
    unsigned short* ctx    = (unsigned short*)(ws + 44040192);           // 8 MB

    // 1) one-time conversion / packing (weights pre-transposed to [N][K])
    cvt_f32_bf16<<<4096, 256, 0, stream>>>(x, x_bf, (int)(M * DMODEL));
    pack_wqkvT<<<1024, 256, 0, stream>>>(Wq, Wk, Wv, wqkvT);
    pack_woT<<<1024, 256, 0, stream>>>(Wo, woT);

    // 2) fused QKV projection: [8192,512] x [512,1536] -> bf16
    gemm_bf16_tn<<<dim3(64, 12), 256, 0, stream>>>(x_bf, wqkvT, qkv,
                                                   (int)M, QKV_LD, DMODEL, 1);

    // 3) transpose V section -> vt[B*H*Dh][S]
    transpose_v<<<dim3(SLEN / 64, BATCH * NHEAD), 256, 0, stream>>>(qkv, vtb);

    // 4) flash attention -> ctx bf16 [8192,512]
    attn_fwd<<<dim3(SLEN / 128, BATCH * NHEAD), 256, 0, stream>>>(qkv, vtb, ctx);

    // 5) output projection: [8192,512] x [512,512] -> fp32 d_out
    gemm_bf16_tn<<<dim3(64, 4), 256, 0, stream>>>(ctx, woT, out,
                                                  (int)M, DMODEL, DMODEL, 0);
}
